// SelfResidualVQ_2259152797751
// MI455X (gfx1250) — compile-verified, matrix operand order checked by reading, not run on hardware
//
#include <hip/hip_runtime.h>
#include <hip/hip_bf16.h>

// CDNA5 / gfx1250: wave32, WMMA bf16 16x16x32, fused residual-VQ.
// Inner GEMM loop: asm-pinned software pipeline (global B distance-3,
// LDS A1 distance-1) with per-operand s_wait_loadcnt/dscnt gating.

typedef __attribute__((ext_vector_type(16))) __bf16   v16bf;
typedef __attribute__((ext_vector_type(8)))  float    v8f;
typedef __attribute__((ext_vector_type(4)))  unsigned u32x4;

#define NTOK   16384
#define DIMD   512
#define NCODE  2048
#define NQ     8
#define NWAVES 4            // waves per block
#define MW     32           // token rows per wave (two 16-row WMMA M-tiles)
#define MTILE  (NWAVES*MW)  // 128 rows per block

struct bfrag { u32x4 lo, hi; };

__device__ __forceinline__ v16bf frag_cast(bfrag f) {
    return __builtin_bit_cast(v16bf, f);
}

// 32B chunk load from global, compile-time byte offset, opaque to the compiler.
template <int BOFF>
__device__ __forceinline__ bfrag ldg_chunk(const unsigned short* base) {
    bfrag f;
    asm volatile("global_load_b128 %0, %2, off offset:%3\n\t"
                 "global_load_b128 %1, %2, off offset:%4"
                 : "=&v"(f.lo), "=&v"(f.hi)
                 : "v"(base), "i"(BOFF), "i"(BOFF + 16));
    return f;
}

// 32B chunk load from LDS (byte address in VGPR + compile-time offset).
template <int BOFF>
__device__ __forceinline__ bfrag ds_chunk(unsigned base) {
    bfrag f;
    asm volatile("ds_load_b128 %0, %2 offset:%3\n\t"
                 "ds_load_b128 %1, %2 offset:%4"
                 : "=&v"(f.lo), "=&v"(f.hi)
                 : "v"(base), "i"(BOFF), "i"(BOFF + 16));
    return f;
}

// Waits bound to the fragment about to be consumed: the WMMA data-depends on
// the "+v" outputs, so it cannot be scheduled above the wait.
__device__ __forceinline__ void wait_b(bfrag& f) {   // <=6: 3 younger B chunks in flight
    asm volatile("s_wait_loadcnt 0x6" : "+v"(f.lo), "+v"(f.hi));
}
__device__ __forceinline__ void wait_a(bfrag& f) {   // <=2: 1 younger A chunk in flight
    asm volatile("s_wait_dscnt 0x2" : "+v"(f.lo), "+v"(f.hi));
}

__device__ __forceinline__ unsigned short f2bf(float f) {
    union { float f; unsigned u; } v; v.f = f;
    unsigned u = v.u;
    unsigned r = u + 0x7FFFu + ((u >> 16) & 1u);   // round-to-nearest-even
    return (unsigned short)(r >> 16);
}

// K index inside a 32-wide chunk for WMMA 16-bit A/B operand layout.
__device__ __forceinline__ int kmapE(int h, int e) {
    return (e < 8) ? (h * 8 + e) : (16 + h * 8 + (e - 8));
}

__global__ void zero_out(float* out, int n) {
    int i = blockIdx.x * blockDim.x + threadIdx.x;
    int stride = gridDim.x * blockDim.x;
    for (; i < n; i += stride) out[i] = 0.0f;
}

__global__ void codebook_norm_pack(const float* __restrict__ cb,
                                   float* __restrict__ cbn,
                                   unsigned short* __restrict__ pB) {
    int n = blockIdx.x * blockDim.x + threadIdx.x;
    if (n >= NCODE) return;
    const float* row = cb + (size_t)n * DIMD;
    float ss = 0.0f;
    for (int k = 0; k < DIMD; ++k) { float v = row[k]; ss += v * v; }
    float scale = 1.0f / fmaxf(sqrtf(ss), 1e-12f);
    int t = n >> 4, nn = n & 15;
    for (int k = 0; k < DIMD; ++k) {
        float v = row[k] * scale;
        cbn[(size_t)n * DIMD + k] = v;
        int c = k >> 5, kk = k & 31;
        int h, e;
        if      (kk < 8)  { h = 0; e = kk; }
        else if (kk < 16) { h = 1; e = kk - 8; }
        else if (kk < 24) { h = 0; e = 8 + (kk - 16); }
        else              { h = 1; e = 8 + (kk - 24); }
        pB[(size_t)(((t * 16 + c) * 32) + h * 16 + nn) * 16 + e] = f2bf(v);
    }
}

// L2-normalize 16 token rows, pack bf16 (WMMA A layout) into the wave's slice.
__device__ __forceinline__ void pack_rows(const float* __restrict__ resin,
                                          int mbase, int lane,
                                          unsigned short* __restrict__ sl) {
    const int row = lane & 15, h = lane >> 4;
    float ss = 0.0f;
    const float* rp = &resin[(size_t)(mbase + row) * DIMD + h * 256];
    for (int i = 0; i < 256; i += 4) {
        float4 v = *(const float4*)&rp[i];
        ss += v.x * v.x + v.y * v.y + v.z * v.z + v.w * v.w;
    }
    ss += __shfl_xor(ss, 16, 32);
    const float scale = 1.0f / fmaxf(sqrtf(ss), 1e-12f);
    const float* rbase = &resin[(size_t)(mbase + row) * DIMD];
    for (int c = 0; c < 16; ++c) {
        unsigned d[8];
#pragma unroll
        for (int p = 0; p < 8; ++p) {
            int K0 = c * 32 + kmapE(h, 2 * p);
            int K1 = c * 32 + kmapE(h, 2 * p + 1);
            d[p] = (unsigned)f2bf(rbase[K0] * scale)
                 | ((unsigned)f2bf(rbase[K1] * scale) << 16);
        }
        int off = (c * 32 + lane) * 16;
        u32x4 lo = {d[0], d[1], d[2], d[3]};
        u32x4 hi = {d[4], d[5], d[6], d[7]};
        *(u32x4*)&sl[off]     = lo;
        *(u32x4*)&sl[off + 8] = hi;
    }
}

// One pipeline step (straight-line macro: template offsets stay constant, the
// register rotation vanishes through SSA renaming).
#define GSTEP(c)                                                               \
    {                                                                          \
        bfrag Bn = ldg_chunk<((c) + 3) * 1024>(gbt);                           \
        bfrag An = ds_chunk<((((c) + 1) & 15)) * 1024>(a_base);                \
        wait_b(B0);                                                            \
        acc0 = __builtin_amdgcn_wmma_f32_16x16x32_bf16(                        \
            false, A0[(c)], false, frag_cast(B0), (short)0, acc0, false, false); \
        wait_a(AQ);                                                            \
        acc1 = __builtin_amdgcn_wmma_f32_16x16x32_bf16(                        \
            false, frag_cast(AQ), false, frag_cast(B0), (short)0, acc1, false, false); \
        B0 = B1; B1 = B2; B2 = Bn; AQ = An;                                    \
    }

__global__ void __launch_bounds__(NWAVES * 32)
vq_step(const float* __restrict__ resin, float* __restrict__ resout,
        const unsigned short* __restrict__ pB,
        const float* __restrict__ cbn,
        float* __restrict__ qsum, float* __restrict__ oidx,
        float* __restrict__ ocommit, int step) {
    __shared__ __align__(32) unsigned short smemA[NWAVES * 16 * 32 * 16]; // 64 KB

    const int lane = threadIdx.x & 31;
    const int w    = threadIdx.x >> 5;
    const int m0   = blockIdx.x * MTILE + w * MW;
    unsigned short* sl = &smemA[w * 8192];

    // ---- tile0: pack to LDS, then hoist into registers ----
    pack_rows(resin, m0, lane, sl);
    v16bf A0[16];
#pragma unroll
    for (int c = 0; c < 16; ++c)
        A0[c] = *(const v16bf*)&sl[(c * 32 + lane) * 16];

    // ---- tile1: pack into the same slice; only read back via opaque asm ----
    pack_rows(resin, m0 + 16, lane, sl);
    asm volatile("" ::: "memory");   // keep tile1 LDS stores alive

    float bv0[8], bv1[8]; int bn0[8], bn1[8];
#pragma unroll
    for (int r = 0; r < 8; ++r) { bv0[r] = bv1[r] = -3.0e38f; bn0[r] = bn1[r] = 0; }

    // per-lane base addresses for asm loads
    const unsigned short* gbt = &pB[(size_t)lane * 16];         // +16KB per t
    unsigned a_base = (unsigned)(size_t)(void*)sl + (unsigned)(lane * 32);

    // pipeline prologue: B chunks 0..2 and A1 chunk 0 in flight
    bfrag B0 = ldg_chunk<0>(gbt);
    bfrag B1 = ldg_chunk<1024>(gbt);
    bfrag B2 = ldg_chunk<2048>(gbt);
    bfrag AQ = ds_chunk<0>(a_base);

    for (int t = 0; t < 128; ++t) {
        v8f acc0 = {}; v8f acc1 = {};
        GSTEP(0)  GSTEP(1)  GSTEP(2)  GSTEP(3)
        GSTEP(4)  GSTEP(5)  GSTEP(6)  GSTEP(7)
        GSTEP(8)  GSTEP(9)  GSTEP(10) GSTEP(11)
        GSTEP(12) GSTEP(13) GSTEP(14) GSTEP(15)

        union { v8f v; float f[8]; } u0, u1; u0.v = acc0; u1.v = acc1;
        int n = t * 16 + (lane & 15);
#pragma unroll
        for (int r = 0; r < 8; ++r) {
            if (u0.f[r] > bv0[r]) { bv0[r] = u0.f[r]; bn0[r] = n; }
            if (u1.f[r] > bv1[r]) { bv1[r] = u1.f[r]; bn1[r] = n; }
        }
        // in-flight asm loads still use gbt: settle translations before bump
        asm volatile("s_wait_xcnt 0x0" : "+v"(gbt));
        gbt += 8192;   // next 16KB N-tile
    }
    // drain: registers of in-flight fragments stay reserved until here
    asm volatile("s_wait_loadcnt 0x0\n\ts_wait_dscnt 0x0"
                 :: "v"(B0.lo), "v"(B0.hi), "v"(B1.lo), "v"(B1.hi),
                    "v"(B2.lo), "v"(B2.hi), "v"(AQ.lo), "v"(AQ.hi));

    // cross-lane argmax within each 16-lane half (tie-break: lower index)
#pragma unroll
    for (int off = 1; off < 16; off <<= 1) {
#pragma unroll
        for (int r = 0; r < 8; ++r) {
            float ov = __shfl_xor(bv0[r], off, 32);
            int   on = __shfl_xor(bn0[r], off, 32);
            if (ov > bv0[r] || (ov == bv0[r] && on < bn0[r])) { bv0[r] = ov; bn0[r] = on; }
            ov = __shfl_xor(bv1[r], off, 32);
            on = __shfl_xor(bn1[r], off, 32);
            if (ov > bv1[r] || (ov == bv1[r] && on < bn1[r])) { bv1[r] = ov; bn1[r] = on; }
        }
    }

    int idxs[32];
#pragma unroll
    for (int r = 0; r < 8; ++r) {
        idxs[r]      = __shfl(bn0[r], 0, 32);   // tile0 rows 0..7
        idxs[8 + r]  = __shfl(bn0[r], 16, 32);  // tile0 rows 8..15
        idxs[16 + r] = __shfl(bn1[r], 0, 32);   // tile1 rows 0..7
        idxs[24 + r] = __shfl(bn1[r], 16, 32);  // tile1 rows 8..15
    }
    if (lane < 16) {
        oidx[(size_t)(m0 + lane) * NQ + step]      = (float)idxs[lane];
        oidx[(size_t)(m0 + 16 + lane) * NQ + step] = (float)idxs[16 + lane];
    }

    // ---- fused update: residual -= q; qsum += q; commit = mean(new_res^2) ----
    float csum = 0.0f;
    for (int rr = 0; rr < MW; ++rr) {
        const int m = m0 + rr;
        const float* qv = &cbn[(size_t)idxs[rr] * DIMD + lane * 16];
        const float* rv = &resin[(size_t)m * DIMD + lane * 16];
        float* ro = &resout[(size_t)m * DIMD + lane * 16];
        float* qs = &qsum[(size_t)m * DIMD + lane * 16];
#pragma unroll
        for (int j = 0; j < 4; ++j) {
            float4 q = ((const float4*)qv)[j];
            float4 r = ((const float4*)rv)[j];
            float4 rn = {r.x - q.x, r.y - q.y, r.z - q.z, r.w - q.w};
            ((float4*)ro)[j] = rn;
            float4 s = ((float4*)qs)[j];
            s.x += q.x; s.y += q.y; s.z += q.z; s.w += q.w;
            ((float4*)qs)[j] = s;
            csum += rn.x * rn.x + rn.y * rn.y + rn.z * rn.z + rn.w * rn.w;
        }
    }
#pragma unroll
    for (int off = 16; off > 0; off >>= 1) csum += __shfl_xor(csum, off, 32);
    if (lane == 0)
        atomicAdd(&ocommit[step], csum * (1.0f / ((float)NTOK * (float)DIMD)));
}

extern "C" void kernel_launch(void* const* d_in, const int* in_sizes, int n_in,
                              void* d_out, int out_size, void* d_ws, size_t ws_size,
                              hipStream_t stream) {
    const float* x  = (const float*)d_in[0];   // (1,16384,512) f32
    const float* cb = (const float*)d_in[1];   // (2048,512)   f32

    float* out     = (float*)d_out;
    float* qsum    = out;                                  // 16384*512
    float* oidx    = out + (size_t)NTOK * DIMD;            // 16384*8
    float* ocommit = oidx + (size_t)NTOK * NQ;             // 8

    char* ws = (char*)d_ws;
    float* resid          = (float*)ws;                                       // 32 MB
    float* cbn            = (float*)(ws + (size_t)NTOK * DIMD * 4);           // 4 MB
    unsigned short* packB = (unsigned short*)(ws + (size_t)NTOK * DIMD * 4
                                                 + (size_t)NCODE * DIMD * 4); // 2 MB
    // prefetch pipeline reads <=3 chunks (3 KB) past packed B; slack lives in ws.

    zero_out<<<2048, 256, 0, stream>>>(out, out_size);
    codebook_norm_pack<<<(NCODE + 255) / 256, 256, 0, stream>>>(cb, cbn, packB);

    for (int s = 0; s < NQ; ++s) {
        const float* rin = (s == 0) ? x : resid;
        vq_step<<<NTOK / MTILE, NWAVES * 32, 0, stream>>>(
            rin, resid, packB, cbn, qsum, oidx, ocommit, s);
    }
}
